// DeltaJANET_27023934227287
// MI455X (gfx1250) — matrix operand order repaired and checked by reading, not compiled, not run on hardware
//
#include <hip/hip_runtime.h>
#include <hip/hip_bf16.h>

// DeltaJANET RNN on gfx1250 (MI455X), telescoped form:
//   pre_t = bias + [h_{t-1} | feats_t] @ [W_hh | W_ih_pad]^T
//   h_t   = (1-sigmoid(pre_f))*sigmoid(pre_g) + sigmoid(pre_f)*h_{t-1}
//   out_t = h_t @ fc_w^T + fc_b
//
// 4 persistent blocks (16 batch rows each), 512 threads = 16 waves (wave32).
// Each wave owns 2 N-tiles (32 of 512 gate columns), weights cached in VGPRs
// as WMMA B-operands; [h|feats] tile lives in LDS as bf16 WMMA A-operand.
// A-operand loads are software-pipelined one K-chunk ahead so the compiler
// can overlap ds_load latency with WMMA issue (dscnt<=2 instead of 0).

#define B_    64
#define T_    2048
#define H_    256
#define IN_   6
#define KTOT  288          // 256 (hidden) + 32 (feats, 6 real + 26 zero)
#define NCH   9            // K chunks of 32
#define HA_S  296          // hA row stride in bf16 elements (592B, 16B aligned)
#define GB_S  260          // gbuf row stride in floats

typedef __attribute__((ext_vector_type(16))) __bf16 v16bf;
typedef __attribute__((ext_vector_type(8)))  __bf16 v8bf;
typedef __attribute__((ext_vector_type(8)))  float  v8f;

__device__ __forceinline__ float sigmoidf_(float v) {
    return 1.0f / (1.0f + __expf(-v));
}

// feats = (i, q, amp, amp^3, q/amp, i/amp) -> bf16 into hA cols 256..261
__device__ __forceinline__ void write_feats(const float* __restrict__ x,
                                            __bf16* __restrict__ hA,
                                            int bblk, int row, int t) {
    const float2 xv = *(const float2*)(x + ((size_t)(bblk + row) * T_ + t) * 2);
    const float ix = xv.x, qx = xv.y;
    const float amp  = __fsqrt_rn(ix * ix + qx * qx);
    const float amp3 = amp * amp * amp;
    const float sn = qx / amp;
    const float cs = ix / amp;
    __bf16* p = hA + row * HA_S + 256;
    p[0] = (__bf16)ix;   p[1] = (__bf16)qx;
    p[2] = (__bf16)amp;  p[3] = (__bf16)amp3;
    p[4] = (__bf16)sn;   p[5] = (__bf16)cs;
}

__global__ __launch_bounds__(512, 1)
void janet_rnn_kernel(const float* __restrict__ x,      // (64,2048,2)
                      const float* __restrict__ w_ih,   // (512,6)
                      const float* __restrict__ w_hh,   // (512,256)
                      const float* __restrict__ b_ih,   // (512,)
                      const float* __restrict__ b_hh,   // (512,)
                      const float* __restrict__ fc_w,   // (2,256)
                      const float* __restrict__ fc_b,   // (2,)
                      float* __restrict__ out)          // (64,2048,2)
{
    __shared__ __bf16 hA[16 * HA_S];      // [h_{t-1} | feats_t] bf16, A-operand
    __shared__ float  gbuf[16 * GB_S];    // sigmoid(pre_g) exchange
    __shared__ float  oacc[16 * 2];       // fused fc partials

    const int tid  = threadIdx.x;
    const int wave = tid >> 5;
    const int lane = tid & 31;
    const int lo16 = lane & 15;
    const int hi   = lane >> 4;           // K-half selector (A/B lane layout)
    const int bblk = blockIdx.x * 16;     // batch base row

    // ---- stage step-invariant B operands (weights) into VGPRs, bf16 ----
    // B[k][n] = W[n][k]; lane holds col n = nbase+tt*16+lo16,
    // K = hi*8+0..7 and 16+hi*8+0..7 (matches A-operand layout).
    v16bf Breg[2][NCH];
    float biasn[2];
    float fw0[2], fw1[2];
    const int nbase = (wave < 8) ? wave * 32 : 256 + (wave - 8) * 32;
    #pragma unroll
    for (int tt = 0; tt < 2; ++tt) {
        const int n = nbase + tt * 16 + lo16;
        biasn[tt] = b_ih[n] + b_hh[n];
        if (wave < 8) {                   // f-waves also need fc rows
            fw0[tt] = fc_w[n];            // fc_w[0][n]
            fw1[tt] = fc_w[H_ + n];       // fc_w[1][n]
        } else {
            fw0[tt] = 0.0f; fw1[tt] = 0.0f;
        }
        #pragma unroll
        for (int kc = 0; kc < 8; ++kc) {  // W_hh chunks
            const int ks = kc * 32 + hi * 8;
            const float* wp = w_hh + (size_t)n * H_ + ks;
            v16bf bv;
            #pragma unroll
            for (int i = 0; i < 16; ++i) bv[i] = (__bf16)wp[i < 8 ? i : i + 8];
            Breg[tt][kc] = bv;
        }
        {                                  // chunk 8: W_ih (K=256..287)
            v16bf bv;
            #pragma unroll
            for (int i = 0; i < 16; ++i) bv[i] = (__bf16)0.0f;
            if (hi == 0) {
                #pragma unroll
                for (int i = 0; i < IN_; ++i) bv[i] = (__bf16)w_ih[n * IN_ + i];
            }
            Breg[tt][8] = bv;
        }
    }

    // Pre-splat bias into persistent v8f accumulator-init registers so the
    // per-step accumulator init is a plain block copy (no re-splat).
    v8f biasv[2];
    #pragma unroll
    for (int tt = 0; tt < 2; ++tt)
        #pragma unroll
        for (int r = 0; r < 8; ++r) biasv[tt][r] = biasn[tt];

    // ---- init LDS: h_0 = 0, feats zero-pad, feats(t=0), oacc=0 ----
    for (int i = tid; i < 16 * HA_S; i += 512) hA[i] = (__bf16)0.0f;
    if (tid < 32) oacc[tid] = 0.0f;
    __syncthreads();
    if (tid < 16) write_feats(x, hA, bblk, tid, 0);
    __syncthreads();

    float hprev[2][8];
    #pragma unroll
    for (int tt = 0; tt < 2; ++tt)
        #pragma unroll
        for (int r = 0; r < 8; ++r) hprev[tt][r] = 0.0f;

    const float fcb0 = fc_b[0], fcb1 = fc_b[1];

    // Per-lane A base: 16-bit A-matrix lane layout: lane holds row m=lo16,
    // K = khalf..khalf+7 and khalf+16..khalf+23 per chunk, khalf = hi*8.
    const __bf16* abase = &hA[lo16 * HA_S + hi * 8];

    for (int t = 0; t < T_; ++t) {
        // ---- pre = bias + [h|feats] @ W^T  (9 bf16 WMMA per tile) ----
        v8f acc[2];
        acc[0] = biasv[0];
        acc[1] = biasv[1];

        // software-pipelined A loads: fetch chunk kc+1 while computing kc
        v8bf a0 = *(const v8bf*)(abase);
        v8bf a1 = *(const v8bf*)(abase + 16);
        #pragma unroll
        for (int kc = 0; kc < NCH; ++kc) {
            v8bf n0 = a0, n1 = a1;
            if (kc + 1 < NCH) {
                n0 = *(const v8bf*)(abase + (kc + 1) * 32);
                n1 = *(const v8bf*)(abase + (kc + 1) * 32 + 16);
            }
            v16bf av;
            #pragma unroll
            for (int i = 0; i < 8; ++i) { av[i] = a0[i]; av[8 + i] = a1[i]; }
            acc[0] = __builtin_amdgcn_wmma_f32_16x16x32_bf16(
                         false, av, false, Breg[0][kc], (short)0, acc[0], false, false);
            acc[1] = __builtin_amdgcn_wmma_f32_16x16x32_bf16(
                         false, av, false, Breg[1][kc], (short)0, acc[1], false, false);
            a0 = n0; a1 = n1;
        }

        // ---- g-waves publish sigmoid(pre_g) ----
        if (wave >= 8) {
            #pragma unroll
            for (int tt = 0; tt < 2; ++tt)
                #pragma unroll
                for (int r = 0; r < 8; ++r) {
                    const int m   = r + 8 * hi;        // C/D layout: M = r + 8*(lane>=16)
                    const int col = (wave - 8) * 32 + tt * 16 + lo16;
                    gbuf[m * GB_S + col] = sigmoidf_(acc[tt][r]);
                }
        }
        __syncthreads();  // barrier1: gates ready, hA reads of step t done

        // ---- f-waves: h_t = (1-f)*g + f*h_{t-1}; write bf16 hA + fc partials ----
        if (wave < 8) {
            #pragma unroll
            for (int r = 0; r < 8; ++r) {
                const int m = r + 8 * hi;
                float o0 = 0.0f, o1 = 0.0f;
                #pragma unroll
                for (int tt = 0; tt < 2; ++tt) {
                    const int col = nbase + tt * 16 + lo16;   // 0..255
                    const float f = sigmoidf_(acc[tt][r]);
                    const float g = gbuf[m * GB_S + col];
                    const float h = (1.0f - f) * g + f * hprev[tt][r];
                    hprev[tt][r] = h;
                    hA[m * HA_S + col] = (__bf16)h;
                    o0 += h * fw0[tt];
                    o1 += h * fw1[tt];
                }
                atomicAdd(&oacc[m * 2 + 0], o0);   // ds_add_f32
                atomicAdd(&oacc[m * 2 + 1], o1);
            }
        }
        // ---- wave 15 stages feats for t+1 (disjoint hA columns) ----
        if (wave == 15 && lane < 16) {
            if (t + 1 < T_) write_feats(x, hA, bblk, lane, t + 1);
            if (t + 2 < T_)
                __builtin_prefetch(x + ((size_t)(bblk + lane) * T_ + (t + 2)) * 2, 0, 1);
        }
        __syncthreads();  // barrier2: hA fully updated for step t+1

        // ---- drain fused fc output, reset accumulators ----
        if (tid < 16) {
            const int m = tid;
            const size_t oidx = ((size_t)(bblk + m) * T_ + t) * 2;
            out[oidx + 0] = oacc[m * 2 + 0] + fcb0;
            out[oidx + 1] = oacc[m * 2 + 1] + fcb1;
            oacc[m * 2 + 0] = 0.0f;
            oacc[m * 2 + 1] = 0.0f;
        }
        // next additions happen only after every thread passes the next
        // barrier1, so the reset is safely ordered.
    }
}

extern "C" void kernel_launch(void* const* d_in, const int* in_sizes, int n_in,
                              void* d_out, int out_size, void* d_ws, size_t ws_size,
                              hipStream_t stream) {
    (void)in_sizes; (void)n_in; (void)d_ws; (void)ws_size; (void)out_size;
    const float* x    = (const float*)d_in[0];
    // d_in[1] = h_0 (reference ignores it: h0 is re-zeroed)
    const float* w_ih = (const float*)d_in[2];
    const float* w_hh = (const float*)d_in[3];
    const float* b_ih = (const float*)d_in[4];
    const float* b_hh = (const float*)d_in[5];
    const float* fc_w = (const float*)d_in[6];
    const float* fc_b = (const float*)d_in[7];
    janet_rnn_kernel<<<dim3(B_ / 16), dim3(512), 0, stream>>>(
        x, w_ih, w_hh, b_ih, b_hh, fc_w, fc_b, (float*)d_out);
}
